// GraphAE_15118284882285
// MI455X (gfx1250) — matrix operand order repaired and checked by reading, not compiled, not run on hardware
//
#include <hip/hip_runtime.h>
#include <hip/hip_bf16.h>
#include <stdint.h>

#define NNODES 20000
#define CIN    1024
#define HID    256
#define NEDGES 160000
#define EPS_GIN 1e-5f
#define EPS_BN  1e-5f

typedef __attribute__((ext_vector_type(16))) __bf16 v16bf;
typedef __attribute__((ext_vector_type(8)))  __bf16 v8bf;
typedef __attribute__((ext_vector_type(8)))  float  v8f;
typedef __attribute__((ext_vector_type(4)))  float  v4f;

// ---------------- elementwise helpers ----------------

// out[i] = s * in[i], float4-vectorized
__global__ void k_scale_f32(const float* __restrict__ in, float* __restrict__ out,
                            float s, int n4) {
    int i = blockIdx.x * blockDim.x + threadIdx.x;
    if (i < n4) {
        v4f v = ((const v4f*)in)[i];
        ((v4f*)out)[i] = v * s;
    }
}

__global__ void k_f32_to_bf16(const float* __restrict__ in, __bf16* __restrict__ out, int n) {
    int i = blockIdx.x * blockDim.x + threadIdx.x;
    if (i < n) out[i] = (__bf16)in[i];
}

// in: [K, C] f32 (row-major)  ->  out: [C, K] bf16 (transposed, row-major)
__global__ void k_transpose_bf16(const float* __restrict__ in, __bf16* __restrict__ out,
                                 int K, int C) {
    int idx = blockIdx.x * blockDim.x + threadIdx.x;
    if (idx < K * C) {
        int k = idx / C, c = idx % C;
        out[(long long)c * K + k] = (__bf16)in[idx];
    }
}

__global__ void k_zero_f32(float* __restrict__ p, int n) {
    int i = blockIdx.x * blockDim.x + threadIdx.x;
    if (i < n) p[i] = 0.0f;
}

// ---------------- GIN scatter-add (one block per edge) ----------------
template<int C>
__global__ void k_scatter_add(const float* __restrict__ feat, float* __restrict__ acc,
                              const long long* __restrict__ src,
                              const long long* __restrict__ dst) {
    int e = blockIdx.x;
    long long s = src[e], d = dst[e];
    const float* fs = feat + s * (long long)C;
    float*       fd = acc  + d * (long long)C;
    for (int c = threadIdx.x; c < C; c += blockDim.x)
        atomicAdd(&fd[c], fs[c]);
}

// ---------------- WMMA fragment helpers (ISA 7.12.2 layouts, wave32) ----------------

// A fragment 16x32 bf16: lane = hi*16 + r holds row M=r;
// vector elems 0..7 -> K = k0 + hi*8 .. +7 ; elems 8..15 -> K = k0 + 16 + hi*8 .. +7
__device__ __forceinline__ v16bf load_a_frag(const __bf16* __restrict__ row, int k0, int hi) {
    v8bf lo = *(const v8bf*)(row + k0 + hi * 8);
    v8bf hh = *(const v8bf*)(row + k0 + 16 + hi * 8);
    return __builtin_shufflevector(lo, hh, 0,1,2,3,4,5,6,7, 8,9,10,11,12,13,14,15);
}

// B fragment 32x16 bf16 from pre-transposed WT[NC, K]: lane holds col N=r;
// elems 0..7 -> K = k0 + hi*8 .. +7 ; elems 8..15 -> K = k0 + 16 + hi*8 .. +7
__device__ __forceinline__ v16bf load_b_frag(const __bf16* __restrict__ row, int k0, int hi) {
    v8bf lo = *(const v8bf*)(row + k0 + hi * 8);
    v8bf hh = *(const v8bf*)(row + k0 + 16 + hi * 8);
    return __builtin_shufflevector(lo, hh, 0,1,2,3,4,5,6,7, 8,9,10,11,12,13,14,15);
}

// C/D 16x16 f32: lane = hi*16 + r -> col N=r ; VGPR j -> row M = hi*8 + j
template<int NC>
__device__ __forceinline__ void store_c_tile(float* __restrict__ Cout, int row0, int col,
                                             const v8f& c, float bv, int hi) {
    float* p = Cout + (long long)(row0 + hi * 8) * NC + col;
    #pragma unroll
    for (int j = 0; j < 8; ++j)
        p[(long long)j * NC] = c[j] + bv;
}

// ---------------- bf16 WMMA GEMM, 32x32 macro-tile (2x2 WMMA tiles) per wave ------
// Cout[NNODES x NC] = A[NNODES x K](bf16) * WT[NC x K](bf16)^T + bias
template<int K, int NC>
__global__ void __launch_bounds__(256)
k_gemm_bf16(const __bf16* __restrict__ A, const __bf16* __restrict__ WT,
            const float* __restrict__ bias, float* __restrict__ Cout) {
    const int wave = threadIdx.x >> 5;
    const int lane = threadIdx.x & 31;
    const int r    = lane & 15;
    const int hi   = lane >> 4;

    const int ntn = NC / 32;                 // macro n-tiles
    const int ntm = NNODES / 32;             // 625
    int tile = blockIdx.x * 8 + wave;
    if (tile >= ntm * ntn) return;           // wave-uniform guard (EXEC stays full)
    int tm = tile / ntn;
    int tn = tile % ntn;

    const __bf16* A0 = A  + (long long)(tm * 32 + r) * K;
    const __bf16* A1 = A0 + (long long)16 * K;
    const __bf16* B0 = WT + (long long)(tn * 32 + r) * K;
    const __bf16* B1 = B0 + (long long)16 * K;

    v8f c00 = {}, c01 = {}, c10 = {}, c11 = {};

    #pragma unroll 2
    for (int k0 = 0; k0 < K; k0 += 32) {
        v16bf a0 = load_a_frag(A0, k0, hi);
        v16bf a1 = load_a_frag(A1, k0, hi);
        v16bf b0 = load_b_frag(B0, k0, hi);
        v16bf b1 = load_b_frag(B1, k0, hi);
        if (k0 + 32 < K) {
            __builtin_prefetch(A0 + k0 + 32, 0, 1);
            __builtin_prefetch(A1 + k0 + 32, 0, 1);
        }
        c00 = __builtin_amdgcn_wmma_f32_16x16x32_bf16(false, a0, false, b0, (short)0, c00, false, false);
        c01 = __builtin_amdgcn_wmma_f32_16x16x32_bf16(false, a0, false, b1, (short)0, c01, false, false);
        c10 = __builtin_amdgcn_wmma_f32_16x16x32_bf16(false, a1, false, b0, (short)0, c10, false, false);
        c11 = __builtin_amdgcn_wmma_f32_16x16x32_bf16(false, a1, false, b1, (short)0, c11, false, false);
    }

    float bv0 = bias[tn * 32 + r];
    float bv1 = bias[tn * 32 + 16 + r];
    store_c_tile<NC>(Cout, tm * 32,      tn * 32 +      r, c00, bv0, hi);
    store_c_tile<NC>(Cout, tm * 32,      tn * 32 + 16 + r, c01, bv1, hi);
    store_c_tile<NC>(Cout, tm * 32 + 16, tn * 32 +      r, c10, bv0, hi);
    store_c_tile<NC>(Cout, tm * 32 + 16, tn * 32 + 16 + r, c11, bv1, hi);
}

// ---------------- batch-norm stats (sum, sumsq per column) ----------------
template<int C>
__global__ void k_bn_stats(const float* __restrict__ h, float* __restrict__ stats,
                           int rows_per_block) {
    int row0 = blockIdx.x * rows_per_block;
    int rend = row0 + rows_per_block; if (rend > NNODES) rend = NNODES;
    for (int c = threadIdx.x; c < C; c += blockDim.x) {
        float s = 0.f, s2 = 0.f;
        for (int rr = row0; rr < rend; ++rr) {
            float v = h[(long long)rr * C + c];
            s += v; s2 += v * v;
        }
        atomicAdd(&stats[c],     s);
        atomicAdd(&stats[C + c], s2);
    }
}

// ---------------- batch-norm apply (+ fused relu / pre-act / (1+eps) init) ---------
template<int C>
__global__ void k_bn_apply(const float* __restrict__ h, const float* __restrict__ stats,
                           const float* __restrict__ gamma, const float* __restrict__ beta,
                           float* __restrict__ out_relu,   // relu(y)
                           float* __restrict__ out_pre,    // y (may alias h; may be null)
                           float* __restrict__ out_agg) {  // (1+eps_gin)*y (may be null)
    long long i = (long long)blockIdx.x * blockDim.x + threadIdx.x;
    if (i >= (long long)NNODES * C) return;
    int c = (int)(i % C);
    const float invN = 1.0f / (float)NNODES;
    float mu    = stats[c] * invN;
    float var   = stats[C + c] * invN - mu * mu;
    float scale = rsqrtf(var + EPS_BN) * gamma[c];
    float y     = (h[i] - mu) * scale + beta[c];
    out_relu[i] = fmaxf(y, 0.0f);
    if (out_pre) out_pre[i] = y;
    if (out_agg) out_agg[i] = (1.0f + EPS_GIN) * y;
}

// ---------------- launch ----------------
extern "C" void kernel_launch(void* const* d_in, const int* in_sizes, int n_in,
                              void* d_out, int out_size, void* d_ws, size_t ws_size,
                              hipStream_t stream) {
    const float*     x      = (const float*)d_in[0];
    const long long* ei     = (const long long*)d_in[1];  // int64 [2, E]
    // d_in[2] = edge_weights (unused by reference)
    const float*     enc_w  = (const float*)d_in[3];      // [1024, 256]
    const float*     enc_b  = (const float*)d_in[4];
    const float*     dec_w  = (const float*)d_in[5];      // [256, 1024]
    const float*     dec_b  = (const float*)d_in[6];
    const float*     gamma1 = (const float*)d_in[7];
    const float*     beta1  = (const float*)d_in[8];
    const float*     gamma2 = (const float*)d_in[9];
    const float*     beta2  = (const float*)d_in[10];

    const long long* src = ei;
    const long long* dst = ei + NEDGES;

    float* z_out  = (float*)d_out;                        // [N, HID]
    float* ae_out = (float*)d_out + (size_t)NNODES * HID; // [N, CIN]

    // ---- workspace carve-up ----
    char*  ws = (char*)d_ws;
    size_t o  = 0;
    float*  bufA_f32  = (float*) (ws + o); o += (size_t)NNODES * CIN * 4;  // hpre, reused as d2
    __bf16* bufA_bf16 = (__bf16*)(ws + o); o += (size_t)NNODES * CIN * 2;  // GEMM1 A
    float*  h1        = (float*) (ws + o); o += (size_t)NNODES * HID * 4;  // GEMM1 out -> h1bn (in-place)
    float*  agg2      = (float*) (ws + o); o += (size_t)NNODES * HID * 4;  // layer-2 aggregation
    __bf16* bufB_bf16 = (__bf16*)(ws + o); o += (size_t)NNODES * HID * 2;  // GEMM2 A
    __bf16* w1t       = (__bf16*)(ws + o); o += (size_t)HID * CIN * 2;     // enc_w^T bf16 [256,1024]
    __bf16* w2t       = (__bf16*)(ws + o); o += (size_t)CIN * HID * 2;     // dec_w^T bf16 [1024,256]
    float*  stats1    = (float*) (ws + o); o += 2 * HID * 4;
    float*  stats2    = (float*) (ws + o); o += 2 * CIN * 4;

    const int T = 256;

    // ---- layer 1: hpre = (1+eps)*x ; += scatter(x) ----
    {
        int n4 = NNODES * CIN / 4;
        k_scale_f32<<<(n4 + T - 1) / T, T, 0, stream>>>(x, bufA_f32, 1.0f + EPS_GIN, n4);
        k_scatter_add<CIN><<<NEDGES, T, 0, stream>>>(x, bufA_f32, src, dst);
        int n = NNODES * CIN;
        k_f32_to_bf16<<<(n + T - 1) / T, T, 0, stream>>>(bufA_f32, bufA_bf16, n);
    }

    // ---- weight prep (transpose + bf16) ----
    k_transpose_bf16<<<(CIN * HID + T - 1) / T, T, 0, stream>>>(enc_w, w1t, CIN, HID);
    k_transpose_bf16<<<(HID * CIN + T - 1) / T, T, 0, stream>>>(dec_w, w2t, HID, CIN);

    // ---- GEMM1: h1 = hpre @ enc_w + enc_b   (20000x1024 x 1024x256) ----
    {
        int tiles = (NNODES / 32) * (HID / 32);   // 625 * 8 = 5000 macro tiles
        k_gemm_bf16<CIN, HID><<<(tiles + 7) / 8, T, 0, stream>>>(bufA_bf16, w1t, enc_b, h1);
    }

    // ---- BN1 + relu -> z ; pre-act kept (in-place) ; agg2 = (1+eps)*h1bn ----
    k_zero_f32<<<(2 * HID + T - 1) / T, T, 0, stream>>>(stats1, 2 * HID);
    k_bn_stats<HID><<<200, T, 0, stream>>>(h1, stats1, 100);
    {
        long long n = (long long)NNODES * HID;
        k_bn_apply<HID><<<(int)((n + T - 1) / T), T, 0, stream>>>(
            h1, stats1, gamma1, beta1, z_out, h1 /*in-place pre-act*/, agg2);
    }

    // ---- layer 2: agg2 += scatter(h1bn) ; to bf16 ----
    k_scatter_add<HID><<<NEDGES, T, 0, stream>>>(h1, agg2, src, dst);
    {
        int n = NNODES * HID;
        k_f32_to_bf16<<<(n + T - 1) / T, T, 0, stream>>>(agg2, bufB_bf16, n);
    }

    // ---- GEMM2: d2 = h2pre @ dec_w + dec_b  (20000x256 x 256x1024), reuse bufA_f32 ----
    {
        int tiles = (NNODES / 32) * (CIN / 32);   // 625 * 32 = 20000 macro tiles
        k_gemm_bf16<HID, CIN><<<(tiles + 7) / 8, T, 0, stream>>>(bufB_bf16, w2t, dec_b, bufA_f32);
    }

    // ---- BN2 + relu -> ae_out ----
    k_zero_f32<<<(2 * CIN + T - 1) / T, T, 0, stream>>>(stats2, 2 * CIN);
    k_bn_stats<CIN><<<200, T, 0, stream>>>(bufA_f32, stats2, 100);
    {
        long long n = (long long)NNODES * CIN;
        k_bn_apply<CIN><<<(int)((n + T - 1) / T), T, 0, stream>>>(
            bufA_f32, stats2, gamma2, beta2, ae_out, nullptr, nullptr);
    }
}
